// QuantumAttention_51049981281138
// MI455X (gfx1250) — matrix operand order repaired and checked by reading, not compiled, not run on hardware
//
#include <hip/hip_runtime.h>

// ---------------------------------------------------------------------------
// QuantumAttention for MI455X (gfx1250, wave32, WMMA bf16 16x16x32, f32 acc)
//
// Reference math folded so every matmul runs on v_wmma_f32_16x16x32_bf16:
//   * mod = cos(phase)*amp folded into projection epilogue
//   * cos_mat(i-j) folded via augmented head dim 128: q~=[q cos a_i, q sin a_i]/8
//   * (w + roll(w,1))/sqrt2 folded into v'[j] = (v[j]+v[j+1 mod S])/sqrt2
// Pipeline: prep(Wt bf16) -> QKV GEMM -> V' transpose -> flash attn -> O GEMM
// Attention stages K~ / V'^T tiles into LDS (double-buffered, async-to-LDS
// global_load_async_to_lds_b128 + s_wait_asynccnt) shared across 8 waves.
// ---------------------------------------------------------------------------

typedef __attribute__((ext_vector_type(16))) __bf16       v16bf;
typedef __attribute__((ext_vector_type(8)))  float        v8f;
typedef __attribute__((ext_vector_type(4)))  unsigned int v4u;
typedef __attribute__((ext_vector_type(4)))  int          v4i;
typedef __attribute__((ext_vector_type(4)))  float        v4f;

#define HID 1024
#define NB  2
#define SEQ 2048
#define NH  16
#define HD  64
#define DA  128              // augmented head dim (cos|sin halves)
#define MROWS (NB * SEQ)     // 4096

#define AS1 __attribute__((address_space(1)))
#define AS3 __attribute__((address_space(3)))

#if defined(__has_builtin)
#if __has_builtin(__builtin_amdgcn_global_load_async_to_lds_b128) && \
    __has_builtin(__builtin_amdgcn_s_wait_asynccnt)
#define USE_ASYNC_LDS 1
#endif
#endif
#ifndef USE_ASYNC_LDS
#define USE_ASYNC_LDS 0
#endif

__device__ __forceinline__ __bf16 f2bf(float f) {
  unsigned u = __builtin_bit_cast(unsigned, f);
  unsigned r = u + 0x7FFFu + ((u >> 16) & 1u);   // round-to-nearest-even
  unsigned short h = (unsigned short)(r >> 16);
  return __builtin_bit_cast(__bf16, h);
}

union FragU { v16bf v; v4u u[2]; __bf16 e[16]; };

__device__ __forceinline__ v8f vzero8() {
  v8f v;
#pragma unroll
  for (int i = 0; i < 8; i++) v[i] = 0.f;
  return v;
}

__device__ __forceinline__ v8f wmma_bf16(v16bf a, v16bf b, v8f c) {
  return __builtin_amdgcn_wmma_f32_16x16x32_bf16(false, a, false, b,
                                                 (short)0, c, false, false);
}

// A fragment (16x32 bf16), ISA 7.12.2 layout: lane row = row0 + lane%16,
// elems 0..7 at K = k0 + 8*(lane/16), elems 8..15 at K = k0 + 16 + 8*(lane/16).
__device__ __forceinline__ v16bf load_fragA(const __bf16* p, int row0, int k0, int ld) {
  int lane = threadIdx.x & 31;
  int lh = lane >> 4;
  const __bf16* b0 = p + (size_t)(row0 + (lane & 15)) * ld + k0 + 8 * lh;
  FragU f;
  f.u[0] = *(const v4u*)(b0);
  f.u[1] = *(const v4u*)(b0 + 16);
  return f.v;
}

// Same A fragment but sourced from f32 data, converted to bf16 in registers.
__device__ __forceinline__ v16bf load_fragA_f32(const float* p, int row0, int k0, int ld) {
  int lane = threadIdx.x & 31;
  int lh = lane >> 4;
  const float* b0 = p + (size_t)(row0 + (lane & 15)) * ld + k0 + 8 * lh;
  v4f a0 = *(const v4f*)(b0);
  v4f a1 = *(const v4f*)(b0 + 4);
  v4f a2 = *(const v4f*)(b0 + 16);
  v4f a3 = *(const v4f*)(b0 + 20);
  FragU f;
#pragma unroll
  for (int i = 0; i < 4; i++) {
    f.e[i]      = f2bf(a0[i]);
    f.e[4 + i]  = f2bf(a1[i]);
    f.e[8 + i]  = f2bf(a2[i]);
    f.e[12 + i] = f2bf(a3[i]);
  }
  return f.v;
}

// B fragment (32x16 bf16) from row-major B^T (T[n][k]): lane col = n0 + lane%16,
// elems 0..7 at K = k0 + 8*(lane/16), elems 8..15 at K = k0 + 16 + 8*(lane/16).
__device__ __forceinline__ v16bf load_fragB(const __bf16* T, int n0, int k0, int ld) {
  int lane = threadIdx.x & 31;
  int lh = lane >> 4;
  const __bf16* b0 = T + (size_t)(n0 + (lane & 15)) * ld + k0 + 8 * lh;
  FragU f;
  f.u[0] = *(const v4u*)(b0);
  f.u[1] = *(const v4u*)(b0 + 16);
  return f.v;
}

// 16-byte global -> LDS copy: async DMA (no VGPR data, ASYNCcnt tracked).
// Builtin signature (from diagnostics): (v4i AS1*, v4i AS3*, imm offset, imm cpol)
__device__ __forceinline__ void copy16_to_lds(const __bf16* g, __bf16* l) {
#if USE_ASYNC_LDS
  __builtin_amdgcn_global_load_async_to_lds_b128(
      (AS1 v4i*)(unsigned long long)g,
      (AS3 v4i*)(unsigned)(unsigned long long)l,
      0, 0);
#else
  *(v4u*)l = *(const v4u*)g;
#endif
}

__device__ __forceinline__ void wait_async_lds() {
#if USE_ASYNC_LDS
  __builtin_amdgcn_s_wait_asynccnt(0);
#endif
}

// ---------------------------------------------------------------------------
// K1: weights f32 -> bf16 transposed (B^T layout), plus mod = cos(phase)*amp
// ---------------------------------------------------------------------------
__global__ void prep_kernel(const float* __restrict__ Wq, const float* __restrict__ Wk,
                            const float* __restrict__ Wv, const float* __restrict__ Wo,
                            const float* __restrict__ phase, const float* __restrict__ amp,
                            __bf16* __restrict__ wt3, __bf16* __restrict__ wot,
                            float* __restrict__ mod) {
  int idx = blockIdx.x * blockDim.x + threadIdx.x;   // 0 .. 1M-1
  int w = blockIdx.y;
  const float* src = (w == 0) ? Wq : (w == 1) ? Wk : (w == 2) ? Wv : Wo;
  __bf16* dst = (w < 3) ? (wt3 + (size_t)w * HID * HID) : wot;
  int k = idx >> 10, n = idx & (HID - 1);
  dst[(size_t)n * HID + k] = f2bf(src[idx]);
  if (w == 0 && idx < NH * HD) mod[idx] = __cosf(phase[idx]) * amp[idx];
}

// ---------------------------------------------------------------------------
// K2: QKV projection GEMM. grid (M/64, N/128, 3); 8 waves: 4 along M, 2 along N.
// Epilogue applies mod and writes augmented Q~/K~ (bf16) or V (bf16).
// ---------------------------------------------------------------------------
__global__ void qkv_gemm_kernel(const float* __restrict__ x, const __bf16* __restrict__ wt3,
                                const float* __restrict__ mod,
                                __bf16* __restrict__ Qa, __bf16* __restrict__ Ka,
                                __bf16* __restrict__ Vbf) {
  int mode = blockIdx.z;                              // 0=Q 1=K 2=V
  const __bf16* wt = wt3 + (size_t)mode * HID * HID;
  int wave = threadIdx.x >> 5, lane = threadIdx.x & 31;
  int lh = lane >> 4, l16 = lane & 15;
  int wm = wave & 3, wn = wave >> 2;
  int mtile = blockIdx.x * 64 + wm * 16;
  int ntile = blockIdx.y * 128 + wn * 64;

  v8f acc[4];
#pragma unroll
  for (int t = 0; t < 4; t++) acc[t] = vzero8();

  for (int kk = 0; kk < HID; kk += 32) {
    v16bf a = load_fragA_f32(x, mtile, kk, HID);
#pragma unroll
    for (int t = 0; t < 4; t++) {
      v16bf b = load_fragB(wt, ntile + t * 16, kk, HID);
      acc[t] = wmma_bf16(a, b, acc[t]);
    }
    __builtin_prefetch(x + (size_t)(mtile + l16) * HID + kk + 64, 0, 1);
  }

#pragma unroll
  for (int t = 0; t < 4; t++) {
#pragma unroll
    for (int r = 0; r < 8; r++) {
      int gm = mtile + r + 8 * lh;                    // C layout: M = r + 8*half
      int gn = ntile + t * 16 + l16;                  // N = lane%16
      int b = gm >> 11, s = gm & (SEQ - 1);
      int h = gn >> 6,  d = gn & 63;
      float v = acc[t][r] * mod[gn];
      if (mode == 2) {
        Vbf[(((size_t)b * NH + h) * SEQ + s) * HD + d] = f2bf(v);
      } else {
        float ang = 6.283185307179586f * (float)s / (float)SEQ;
        float sn, cs;
        __sincosf(ang, &sn, &cs);
        float scale = (mode == 0) ? 0.125f : 1.0f;    // fold 1/sqrt(D) into Q~
        __bf16* dst = (mode == 0) ? Qa : Ka;
        size_t base = (((size_t)b * NH + h) * SEQ + s) * DA + d;
        dst[base]      = f2bf(v * cs * scale);
        dst[base + HD] = f2bf(v * sn * scale);
      }
    }
  }
}

// ---------------------------------------------------------------------------
// K3: V'^T[d][j] = (V[j][d] + V[(j+1)%S][d]) / sqrt(2)   (per b,h)
// ---------------------------------------------------------------------------
__global__ void make_vt_kernel(const __bf16* __restrict__ Vbf, __bf16* __restrict__ Vt) {
  int idx = blockIdx.x * blockDim.x + threadIdx.x;    // B*H*S*D = 4M
  int bh = idx >> 17;                                 // / (SEQ*HD)
  int rem = idx & (SEQ * HD - 1);
  int j = rem >> 6, d = rem & 63;
  float v0 = (float)Vbf[((size_t)bh * SEQ + j) * HD + d];
  float v1 = (float)Vbf[((size_t)bh * SEQ + ((j + 1) & (SEQ - 1))) * HD + d];
  Vt[((size_t)bh * HD + d) * SEQ + j] = f2bf((v0 + v1) * 0.70710678118f);
}

// ---------------------------------------------------------------------------
// K4: flash attention. grid (S/128, H, B); 8 waves, one 16-row Q tile each.
// K~ 32x128 and V'^T 64x32 tiles are staged into LDS per j-step, shared by all
// 8 waves, double-buffered; async global->LDS DMA overlaps the WMMA compute.
// Online softmax; row M = r + 8*(lane/16) lives in one 16-lane half -> shfl_xor.
// ---------------------------------------------------------------------------
__device__ __forceinline__ void stage_kv(const __bf16* __restrict__ k,
                                         const __bf16* __restrict__ vt,
                                         __bf16* kb, __bf16* vb, int j0) {
  int tid = threadIdx.x;
  // K~ tile: 32 rows x 128 cols bf16 = 512 x 16B chunks, 2 per thread
#pragma unroll
  for (int i = 0; i < 2; i++) {
    int c = tid * 2 + i;
    int row = c >> 4, off = (c & 15) * 8;
    copy16_to_lds(k + (size_t)(j0 + row) * DA + off, kb + row * DA + off);
  }
  // V'^T tile: 64 rows x 32 cols bf16 = 256 x 16B chunks, 1 per thread
  {
    int row = tid >> 2, off = (tid & 3) * 8;
    copy16_to_lds(vt + (size_t)row * SEQ + j0 + off, vb + row * 32 + off);
  }
}

__global__ void attn_kernel(const __bf16* __restrict__ Qa, const __bf16* __restrict__ Ka,
                            const __bf16* __restrict__ Vt, __bf16* __restrict__ Obf) {
  __shared__ __bf16 kbuf[2][32 * DA];                 // 2 x 8 KB
  __shared__ __bf16 vbuf[2][HD * 32];                 // 2 x 4 KB
  __shared__ __bf16 pbuf[8][16 * 32];                 // 8 KB, one 16x32 tile / wave
  int wave = threadIdx.x >> 5, lane = threadIdx.x & 31;
  int lh = lane >> 4, l16 = lane & 15;
  int b = blockIdx.z, h = blockIdx.y;
  size_t bh = (size_t)b * NH + h;
  const __bf16* q  = Qa + bh * SEQ * DA;
  const __bf16* k  = Ka + bh * SEQ * DA;
  const __bf16* vt = Vt + bh * HD * SEQ;
  int i0 = blockIdx.x * 128 + wave * 16;

  v16bf qf[4];
#pragma unroll
  for (int ks = 0; ks < 4; ks++) qf[ks] = load_fragA(q, i0, ks * 32, DA);

  v8f acco[4];
#pragma unroll
  for (int t = 0; t < 4; t++) acco[t] = vzero8();
  float mrow[8], lrow[8];
#pragma unroll
  for (int r = 0; r < 8; r++) { mrow[r] = -3.0e30f; lrow[r] = 0.f; }

  __bf16* pb = &pbuf[wave][0];

  stage_kv(k, vt, &kbuf[0][0], &vbuf[0][0], 0);       // prologue: buffer 0

  for (int jt = 0; jt < SEQ / 32; jt++) {
    int cur = jt & 1;
    int j0 = jt * 32;
    wait_async_lds();                                 // my async writes landed
    __syncthreads();                                  // tile[cur] ready, tile[1-cur] free
    if (jt + 1 < SEQ / 32)
      stage_kv(k, vt, &kbuf[1 - cur][0], &vbuf[1 - cur][0], j0 + 32);

    const __bf16* kb = &kbuf[cur][0];
    const __bf16* vb = &vbuf[cur][0];

    v8f s0 = vzero8(), s1 = vzero8();
#pragma unroll
    for (int ks = 0; ks < 4; ks++) {                  // K~ rows are B^T layout
      v16bf kb0 = load_fragB(kb, 0,  ks * 32, DA);
      v16bf kb1 = load_fragB(kb, 16, ks * 32, DA);
      s0 = wmma_bf16(qf[ks], kb0, s0);
      s1 = wmma_bf16(qf[ks], kb1, s1);
    }
    float corr[8];
#pragma unroll
    for (int r = 0; r < 8; r++) {
      float vmax = fmaxf(s0[r], s1[r]);
#pragma unroll
      for (int mm = 1; mm < 16; mm <<= 1) vmax = fmaxf(vmax, __shfl_xor(vmax, mm, 32));
      float mnew = fmaxf(mrow[r], vmax);
      float c  = __expf(mrow[r] - mnew);
      float p0 = __expf(s0[r] - mnew);
      float p1 = __expf(s1[r] - mnew);
      float rs = p0 + p1;
#pragma unroll
      for (int mm = 1; mm < 16; mm <<= 1) rs += __shfl_xor(rs, mm, 32);
      lrow[r] = lrow[r] * c + rs;
      mrow[r] = mnew;
      corr[r] = c;
      s0[r] = p0;
      s1[r] = p1;
    }
#pragma unroll
    for (int t = 0; t < 4; t++)
#pragma unroll
      for (int r = 0; r < 8; r++) acco[t][r] *= corr[r];

    // accumulator(C layout) -> LDS bf16 P tile (row-major 16x32), wave-private
#pragma unroll
    for (int r = 0; r < 8; r++) {
      int row = r + 8 * lh;
      pb[row * 32 + l16]      = f2bf(s0[r]);
      pb[row * 32 + 16 + l16] = f2bf(s1[r]);
    }
    v16bf pf = load_fragA(pb, 0, 0, 32);              // ds_load_b128 x2
#pragma unroll
    for (int t = 0; t < 4; t++) {
      v16bf vfr = load_fragB(vb, t * 16, 0, 32);      // V'^T tile in LDS
      acco[t] = wmma_bf16(pf, vfr, acco[t]);
    }
  }

  float inv[8];
#pragma unroll
  for (int r = 0; r < 8; r++) inv[r] = 1.0f / lrow[r];
#pragma unroll
  for (int t = 0; t < 4; t++)
#pragma unroll
    for (int r = 0; r < 8; r++) {
      int s = i0 + r + 8 * lh;
      int col = h * HD + t * 16 + l16;
      Obf[((size_t)b * SEQ + s) * HID + col] = f2bf(acco[t][r] * inv[r]);
    }
}

// ---------------------------------------------------------------------------
// K5: out = O @ Wo (bf16 WMMA, f32 out). grid (M/64, N/128).
// ---------------------------------------------------------------------------
__global__ void out_gemm_kernel(const __bf16* __restrict__ Obf, const __bf16* __restrict__ wot,
                                float* __restrict__ out) {
  int wave = threadIdx.x >> 5, lane = threadIdx.x & 31;
  int lh = lane >> 4, l16 = lane & 15;
  int wm = wave & 3, wn = wave >> 2;
  int mtile = blockIdx.x * 64 + wm * 16;
  int ntile = blockIdx.y * 128 + wn * 64;

  v8f acc[4];
#pragma unroll
  for (int t = 0; t < 4; t++) acc[t] = vzero8();

  for (int kk = 0; kk < HID; kk += 32) {
    v16bf a = load_fragA(Obf, mtile, kk, HID);
#pragma unroll
    for (int t = 0; t < 4; t++) {
      v16bf b = load_fragB(wot, ntile + t * 16, kk, HID);
      acc[t] = wmma_bf16(a, b, acc[t]);
    }
    __builtin_prefetch(Obf + (size_t)(mtile + l16) * HID + kk + 64, 0, 1);
  }
#pragma unroll
  for (int t = 0; t < 4; t++)
#pragma unroll
    for (int r = 0; r < 8; r++)
      out[(size_t)(mtile + r + 8 * lh) * HID + ntile + t * 16 + l16] = acc[t][r];
}

// ---------------------------------------------------------------------------
extern "C" void kernel_launch(void* const* d_in, const int* in_sizes, int n_in,
                              void* d_out, int out_size, void* d_ws, size_t ws_size,
                              hipStream_t stream) {
  const float* x  = (const float*)d_in[0];
  const float* Wq = (const float*)d_in[1];
  const float* Wk = (const float*)d_in[2];
  const float* Wv = (const float*)d_in[3];
  const float* Wo = (const float*)d_in[4];
  const float* ph = (const float*)d_in[5];
  const float* am = (const float*)d_in[6];
  float* out = (float*)d_out;

  // workspace layout (~64 MB total)
  char* ws = (char*)d_ws;
  size_t off = 0;
  __bf16* wt3 = (__bf16*)(ws + off); off += (size_t)3 * HID * HID * 2;   // 6 MB
  __bf16* wot = (__bf16*)(ws + off); off += (size_t)HID * HID * 2;       // 2 MB
  float*  mod = (float*)(ws + off);  off += (size_t)NH * HD * 4;         // 4 KB
  off = (off + 255) & ~(size_t)255;
  __bf16* Qa  = (__bf16*)(ws + off); off += (size_t)NB * NH * SEQ * DA * 2;  // 16 MB
  __bf16* Ka  = (__bf16*)(ws + off); off += (size_t)NB * NH * SEQ * DA * 2;  // 16 MB
  __bf16* Vbf = (__bf16*)(ws + off); off += (size_t)NB * NH * SEQ * HD * 2;  // 8 MB
  __bf16* Vt  = (__bf16*)(ws + off); off += (size_t)NB * NH * HD * SEQ * 2;  // 8 MB
  __bf16* Obf = (__bf16*)(ws + off); off += (size_t)MROWS * HID * 2;         // 8 MB
  (void)ws_size; (void)in_sizes; (void)n_in; (void)out_size;

  prep_kernel<<<dim3(HID * HID / 256, 4), 256, 0, stream>>>(Wq, Wk, Wv, Wo, ph, am,
                                                            wt3, wot, mod);
  qkv_gemm_kernel<<<dim3(MROWS / 64, HID / 128, 3), 256, 0, stream>>>(x, wt3, mod,
                                                                      Qa, Ka, Vbf);
  make_vt_kernel<<<(NB * NH * SEQ * HD) / 256, 256, 0, stream>>>(Vbf, Vt);
  attn_kernel<<<dim3(SEQ / 128, NH, NB), 256, 0, stream>>>(Qa, Ka, Vt, Obf);
  out_gemm_kernel<<<dim3(MROWS / 64, HID / 128), 256, 0, stream>>>(Obf, wot, out);
}
